// SSIM3D_77962246357410
// MI455X (gfx1250) — compile-verified, matrix operand order checked by reading, not run on hardware
//
#include <hip/hip_runtime.h>
#include <hip/hip_bf16.h>

// ---------------------------------------------------------------------------
// Fused 3-D SSIM for 4x1x128^3 fp32 volumes on gfx1250 (MI455X).
//
// - Halo tiles staged to LDS with GLOBAL_LOAD_ASYNC_TO_LDS_B32 (ASYNCcnt).
// - Separable 11-tap Gaussian as banded-matrix V_WMMA_F32_16X16X32_F16:
//     D(16x16) = A(16x32 band) x B(32x16), f16 in / f32 accumulate.
// - All K-rows padded to 32 elements (zero pad) so every B fragment is
//   2x (f16) or 4x (f32) ds_load_b128, no per-element bounds checks.
// - 3 passes (x,y,z) over 5 fields (I1, I2, I1^2, I2^2, I1*I2), then
//   pointwise SSIM + wave32 shuffle reduction + one f32 atomic per wave.
// - One workgroup (8 wave32) per 16^3 output tile; 299 KiB LDS < 320 KiB.
// ---------------------------------------------------------------------------

typedef __attribute__((ext_vector_type(16))) _Float16 v16h;
typedef __attribute__((ext_vector_type(8)))  _Float16 v8h;
typedef __attribute__((ext_vector_type(8)))  float    v8f;
typedef __attribute__((ext_vector_type(4)))  float    v4f;

#define VOL   128
#define TILE  16
#define HALO  26            // TILE + 11 - 1
#define PAD   5
#define KPAD  32            // padded K-row length

#define A_ELEMS   (HALO*HALO*KPAD)        // 21632 f32 per image (rows padded)
#define A1_OFF    0u                      // byte offsets inside dynamic LDS
#define A2_OFF    (A_ELEMS*4u)            // 86528
#define SX_OFF    (2u*A_ELEMS*4u)         // 173056
#define SX_ELEMS  (5*HALO*TILE*KPAD)      // 66560 halves
#define LDS_BYTES (SX_OFF + SX_ELEMS*2u)  // 306176 bytes (299 KiB)

// Normalized 1-D Gaussian, sigma = 1.5, size 11 (matches reference).
__device__ __constant__ float c_W[11] = {
    0.00102838f, 0.00759876f, 0.03600077f, 0.10936060f, 0.21300510f,
    0.26601171f, 0.21300510f, 0.10936060f, 0.03600077f, 0.00759876f,
    0.00102838f
};

__device__ inline v8f wmma_blur(v16h a, v16h b, v8f c) {
    // (neg_a, A, neg_b, B, c_mod, C, reuse_a, reuse_b)
    return __builtin_amdgcn_wmma_f32_16x16x32_f16(false, a, false, b,
                                                  (short)0, c, false, false);
}

extern "C" __global__ void ssim3d_zero_out(float* out) {
    if (threadIdx.x == 0) out[0] = 0.0f;
}

extern "C" __global__ __launch_bounds__(256)
void ssim3d_fused_wmma(const float* __restrict__ img1,
                       const float* __restrict__ img2,
                       float* __restrict__ out)
{
    extern __shared__ __align__(16) char smem_raw[];
    float*    A1f = reinterpret_cast<float*>(smem_raw);            // [z][y][x32] f32
    float*    A2f = A1f + A_ELEMS;
    _Float16* Sx  = reinterpret_cast<_Float16*>(smem_raw + SX_OFF);// [q][z][x][y32] f16
    _Float16* Sy  = reinterpret_cast<_Float16*>(smem_raw);         // [q][y][x][z32] f16
                                                                   // overlays A after pass 1
    const int tid  = threadIdx.x;
    const int wid  = tid >> 5;        // 8 waves
    const int lane = tid & 31;
    const int mrow = lane & 15;       // M (output row) within fragment
    const int hi   = lane >> 4;       // lane half selects K sub-block
    const int ncol = lane & 15;       // N (column) within fragment

    // Tile decode: 4 batches * 8^3 tiles.
    const int bidx = blockIdx.x;
    const int b  = bidx >> 9;
    const int tz = (bidx >> 6) & 7;
    const int ty = (bidx >> 3) & 7;
    const int tx = bidx & 7;
    const int oz = tz * TILE - PAD, oy = ty * TILE - PAD, ox = tx * TILE - PAD;
    const long gbase = (long)b * VOL * VOL * VOL;   // channel == 1

    // ---- Stage 0a: zero the A region (covers x-pad and OOB halo cells) ----
    {
        v4f* az = reinterpret_cast<v4f*>(smem_raw);
        const int nvec = (2 * A_ELEMS) / 4;          // 10816 x 16B
        for (int i = tid; i < nvec; i += 256) az[i] = (v4f){0.f, 0.f, 0.f, 0.f};
    }
    __syncthreads();

    // ---- Stage 0b: async halo-tile load straight into LDS (f32) -----------
    for (int idx = tid; idx < HALO*HALO*HALO; idx += 256) {
        int lz = idx / (HALO*HALO);
        int rem = idx % (HALO*HALO);
        int ly = rem / HALO, lx = rem % HALO;
        int gz = oz + lz, gy = oy + ly, gx = ox + lx;
        if ((unsigned)gz < (unsigned)VOL && (unsigned)gy < (unsigned)VOL &&
            (unsigned)gx < (unsigned)VOL) {
            long g = gbase + ((long)gz * VOL + gy) * VOL + gx;
            unsigned ai = ((unsigned)(lz * HALO + ly)) * KPAD + (unsigned)lx;
            unsigned l1 = A1_OFF + 4u * ai;
            unsigned l2 = A2_OFF + 4u * ai;
            unsigned long long g1 = (unsigned long long)(img1 + g);
            unsigned long long g2 = (unsigned long long)(img2 + g);
            asm volatile("global_load_async_to_lds_b32 %0, %1, off"
                         :: "v"(l1), "v"(g1) : "memory");
            asm volatile("global_load_async_to_lds_b32 %0, %1, off"
                         :: "v"(l2), "v"(g2) : "memory");
        }
    }
    asm volatile("s_wait_asynccnt 0x0" ::: "memory");
    __syncthreads();

    // ---- Banded Gaussian A fragment (16x32, f16 layout per ISA 7.12.2) ----
    // lane<16 holds K=0..7,16..23 ; lane>=16 holds K=8..15,24..31 (2/half-VGPR)
    v16h Af;
#pragma unroll
    for (int j = 0; j < 16; ++j) {
        int K = ((j < 8) ? j : (j + 8)) + hi * 8;
        int d = K - mrow;
        Af[j] = (_Float16)((d >= 0 && d <= 10) ? c_W[d] : 0.0f);
    }

    // f32 K-row -> f16 B fragment (4x ds_load_b128 + converts).
    auto loadB_f32 = [&](const float* row) -> v16h {
        v4f a0 = *(const v4f*)(row + hi*8);
        v4f a1 = *(const v4f*)(row + hi*8 + 4);
        v4f a2 = *(const v4f*)(row + 16 + hi*8);
        v4f a3 = *(const v4f*)(row + 20 + hi*8);
        v16h B;
#pragma unroll
        for (int e = 0; e < 4; ++e) {
            B[e]      = (_Float16)a0[e];
            B[e + 4]  = (_Float16)a1[e];
            B[e + 8]  = (_Float16)a2[e];
            B[e + 12] = (_Float16)a3[e];
        }
        return B;
    };

    // f16 K-row -> B fragment (2x ds_load_b128, zero pad already in LDS).
    auto loadB_f16 = [&](const _Float16* row) -> v16h {
        v8h lo = *(const v8h*)(row + hi*8);
        v8h hh = *(const v8h*)(row + 16 + hi*8);
        return __builtin_shufflevector(lo, hh, 0,1,2,3,4,5,6,7,
                                               8,9,10,11,12,13,14,15);
    };

    // ---- Pass 1: blur along x. 26 z-planes * 2 y-subtiles = 52 fragments --
    for (int frag = wid; frag < 52; frag += 8) {
        int z    = frag >> 1;
        int yt   = frag & 1;
        int ycol = yt * 16 + ncol;            // < 32 always
        bool cval = (ycol < HALO);
        int  ycl  = cval ? ycol : 0;          // invalid columns are discarded

        v16h B0 = loadB_f32(&A1f[(long)(z * HALO + ycl) * KPAD]);
        v16h B1 = loadB_f32(&A2f[(long)(z * HALO + ycl) * KPAD]);
        v16h B2 = B0 * B0;                    // v_pk_mul_f16
        v16h B3 = B1 * B1;
        v16h B4 = B0 * B1;

        v8f z8 = {};
        v8f c0 = wmma_blur(Af, B0, z8);
        v8f c1 = wmma_blur(Af, B1, z8);
        v8f c2 = wmma_blur(Af, B2, z8);
        v8f c3 = wmma_blur(Af, B3, z8);
        v8f c4 = wmma_blur(Af, B4, z8);

        // Store D; invalid columns write zeros -> Sx pad stays clean.
        const long sq = (long)HALO * TILE * KPAD;
#pragma unroll
        for (int r = 0; r < 8; ++r) {
            int x = r + hi * 8;               // D layout: M = r + 8*(lane>=16)
            long o = ((long)z * TILE + x) * KPAD + ycol;
            Sx[o + 0 * sq] = cval ? (_Float16)c0[r] : (_Float16)0.0f;
            Sx[o + 1 * sq] = cval ? (_Float16)c1[r] : (_Float16)0.0f;
            Sx[o + 2 * sq] = cval ? (_Float16)c2[r] : (_Float16)0.0f;
            Sx[o + 3 * sq] = cval ? (_Float16)c3[r] : (_Float16)0.0f;
            Sx[o + 4 * sq] = cval ? (_Float16)c4[r] : (_Float16)0.0f;
        }
    }
    __syncthreads();

    // Zero Sy z-pad (halves 26..31 of each row); no race with pass-2 stores.
    for (int i = tid; i < 5 * TILE * TILE; i += 256) {
        unsigned* pz = reinterpret_cast<unsigned*>(Sy + (long)i * KPAD + HALO);
        pz[0] = 0u; pz[1] = 0u; pz[2] = 0u;
    }

    // ---- Pass 2: blur along y. 26 z-plane fragments, 5 fields each --------
    for (int fz = wid; fz < HALO; fz += 8) {
#pragma unroll
        for (int q = 0; q < 5; ++q) {
            v16h Bv = loadB_f16(&Sx[(((long)q * HALO + fz) * TILE + ncol) * KPAD]);
            v8f z8 = {};
            v8f c = wmma_blur(Af, Bv, z8);
#pragma unroll
            for (int r = 0; r < 8; ++r) {
                int y = r + hi * 8;
                Sy[(((long)q * TILE + y) * TILE + ncol) * KPAD + fz] = (_Float16)c[r];
            }
        }
    }
    __syncthreads();

    // ---- Pass 3: blur along z, then pointwise SSIM + reduction ------------
    const float C1 = 1e-4f, C2 = 9e-4f;
    float lsum = 0.0f;
    for (int fy = wid; fy < 16; fy += 8) {
        v8f z8 = {};
        v8f c0 = wmma_blur(Af, loadB_f16(&Sy[(((long)0*TILE + fy)*TILE + ncol)*KPAD]), z8);
        v8f c1 = wmma_blur(Af, loadB_f16(&Sy[(((long)1*TILE + fy)*TILE + ncol)*KPAD]), z8);
        v8f c2 = wmma_blur(Af, loadB_f16(&Sy[(((long)2*TILE + fy)*TILE + ncol)*KPAD]), z8);
        v8f c3 = wmma_blur(Af, loadB_f16(&Sy[(((long)3*TILE + fy)*TILE + ncol)*KPAD]), z8);
        v8f c4 = wmma_blur(Af, loadB_f16(&Sy[(((long)4*TILE + fy)*TILE + ncol)*KPAD]), z8);
#pragma unroll
        for (int r = 0; r < 8; ++r) {
            float mu1 = c0[r], mu2 = c1[r];
            float mu1s = mu1 * mu1, mu2s = mu2 * mu2, m12 = mu1 * mu2;
            float s1  = c2[r] - mu1s;
            float s2  = c3[r] - mu2s;
            float s12 = c4[r] - m12;
            float num = (2.0f * m12 + C1) * (2.0f * s12 + C2);
            float den = (mu1s + mu2s + C1) * (s1 + s2 + C2);
            lsum += num / den;
        }
    }

    // wave32 reduction, then one atomic per wave (pre-scaled by 1/N).
#pragma unroll
    for (int off = 16; off > 0; off >>= 1)
        lsum += __shfl_down(lsum, off, 32);
    if (lane == 0)
        atomicAdd(out, lsum * (1.0f / 8388608.0f));   // N = 4*128^3
}

extern "C" void kernel_launch(void* const* d_in, const int* in_sizes, int n_in,
                              void* d_out, int out_size, void* d_ws, size_t ws_size,
                              hipStream_t stream)
{
    const float* img1 = (const float*)d_in[0];
    const float* img2 = (const float*)d_in[1];
    float* out = (float*)d_out;

    hipLaunchKernelGGL(ssim3d_zero_out, dim3(1), dim3(32), 0, stream, out);

    hipLaunchKernelGGL(ssim3d_fused_wmma, dim3(4 * 8 * 8 * 8), dim3(256),
                       LDS_BYTES, stream, img1, img2, out);
}
